// DecoderBlock_26396869001657
// MI455X (gfx1250) — matrix operand order repaired and checked by reading, not compiled, not run on hardware
//
#include <hip/hip_runtime.h>
#include <math.h>

typedef _Float16 half_t;
typedef __attribute__((ext_vector_type(8)))  _Float16 v8h;
typedef __attribute__((ext_vector_type(16))) _Float16 v16h;
typedef __attribute__((ext_vector_type(8)))  float    v8f;

#define EPSV 1e-5f

static constexpr int BN = 512;     // B*N
static constexpr int TT = 64;      // time after upsample
static constexpr int DM = 128;     // d_model
static constexpr int DI = 256;     // d_inner
static constexpr int M_ROWS = BN * TT; // 32768

// ---------------------------------------------------------------------------
// WMMA GEMM:  C[M,Npad] = A[M,K] * WT[K,Npad] (+bias) (opt relu)
// A : f16 row-major, leading dim lda (16B-aligned rows)
// WT: f16, K x Npad, transposed+padded weights (pad cols are zero)
// One wave computes a 32x16 C block (two 16x16 M-tiles sharing B fragment).
// K must be a multiple of 32; Npad a multiple of 16; M a multiple of 32.
// ---------------------------------------------------------------------------
__global__ __launch_bounds__(32) void k_gemm_wmma(
    const half_t* __restrict__ A, int lda,
    const half_t* __restrict__ WT,
    const float* __restrict__ bias,
    float* __restrict__ C, int ldc,
    int Npad, int K, int relu)
{
    const int m0 = blockIdx.x * 32;
    const int n0 = blockIdx.y * 16;
    const int lane = threadIdx.x;          // 0..31
    const int row  = lane & 15;
    const int grp  = lane >> 4;

    v8f acc0 = {};
    v8f acc1 = {};

    const half_t* Ap0 = A + (size_t)(m0 + row) * lda + 8 * grp;
    const half_t* Ap1 = Ap0 + (size_t)16 * lda;
    const half_t* Bp  = WT + (size_t)n0 + row;

    #pragma unroll 2
    for (int k0 = 0; k0 < K; k0 += 32) {
        // A fragments: ISA 16-bit A layout == two contiguous 8-half chunks
        v16h a0, a1;
        *(v8h*)&a0       = *(const v8h*)(Ap0 + k0);
        *((v8h*)&a0 + 1) = *(const v8h*)(Ap0 + k0 + 16);
        *(v8h*)&a1       = *(const v8h*)(Ap1 + k0);
        *((v8h*)&a1 + 1) = *(const v8h*)(Ap1 + k0 + 16);
        // B fragment: lane column n = lane&15, K in same order as A
        v16h b;
        #pragma unroll
        for (int s = 0; s < 8; ++s) {
            b[s]     = Bp[(size_t)(k0 + 8 * grp + s) * Npad];
            b[s + 8] = Bp[(size_t)(k0 + 16 + 8 * grp + s) * Npad];
        }

        acc0 = __builtin_amdgcn_wmma_f32_16x16x32_f16(
            false, a0, false, b, (short)0, acc0, false, false);
        acc1 = __builtin_amdgcn_wmma_f32_16x16x32_f16(
            false, a1, false, b, (short)0, acc1, false, false);
    }

    // C layout: lane holds column n = lane&15 ; VGPR v holds row m = v + 8*grp
    const int n = n0 + row;
    const float bv = bias ? bias[n] : 0.0f;
    float* Cp0 = C + (size_t)(m0 + 8 * grp) * ldc + n;
    float* Cp1 = Cp0 + (size_t)16 * ldc;
    #pragma unroll
    for (int v = 0; v < 8; ++v) {
        float v0 = acc0[v] + bv;
        float v1 = acc1[v] + bv;
        if (relu) { v0 = fmaxf(v0, 0.0f); v1 = fmaxf(v1, 0.0f); }
        Cp0[(size_t)v * ldc] = v0;
        Cp1[(size_t)v * ldc] = v1;
    }
}

// ---------------------------------------------------------------------------
// Stage weights: W[N,K] f32 -> WT[K,Npad] f16 (transposed, zero-padded)
// ---------------------------------------------------------------------------
__global__ void k_stage_wT(const float* __restrict__ src, half_t* __restrict__ dst,
                           int N, int K, int Npad)
{
    int total = K * Npad;
    for (int i = blockIdx.x * blockDim.x + threadIdx.x; i < total;
         i += gridDim.x * blockDim.x) {
        int k = i / Npad;
        int n = i - k * Npad;
        dst[i] = (n < N) ? (half_t)src[(size_t)n * K + k] : (half_t)0.0f;
    }
}

// f32 -> f16 conversion (grid-stride)
__global__ void k_tof16(const float* __restrict__ src, half_t* __restrict__ dst, size_t n)
{
    for (size_t i = (size_t)blockIdx.x * blockDim.x + threadIdx.x; i < n;
         i += (size_t)gridDim.x * blockDim.x)
        dst[i] = (half_t)src[i];
}

// ---------------------------------------------------------------------------
// Transposed-conv upsample (stride-2, K=4) + stage concat(xu, skip) as f16
// hS1 row (bn*64+t): cols [0,128)=xu channel c, cols [128,256)=skip[bn,t,c]
// ---------------------------------------------------------------------------
__global__ void k_upsample_stage(const float* __restrict__ x,
                                 const float* __restrict__ skip,
                                 const float* __restrict__ up_w,
                                 const float* __restrict__ up_b,
                                 half_t* __restrict__ hS1)
{
    int idx = blockIdx.x * blockDim.x + threadIdx.x;   // over M*128
    if (idx >= M_ROWS * DM) return;
    int c = idx & 127;
    int rowi = idx >> 7;
    int t  = rowi & 63;
    int bn = rowi >> 6;

    int j0, w0sel, j1, w1sel;
    if ((t & 1) == 0) { j0 = t / 2 - 1;   w0sel = 3; j1 = t / 2;       w1sel = 1; }
    else              { j0 = (t - 1) / 2; w0sel = 2; j1 = (t + 1) / 2; w1sel = 0; }

    const float* xb = x + (size_t)bn * 32 * DM;
    float acc = up_b[c];
    const bool v0 = (j0 >= 0 && j0 < 32);
    const bool v1 = (j1 >= 0 && j1 < 32);
    for (int i = 0; i < DM; ++i) {
        float w0 = up_w[(i * DM + c) * 4 + w0sel];
        float w1 = up_w[(i * DM + c) * 4 + w1sel];
        float x0 = v0 ? xb[j0 * DM + i] : 0.0f;
        float x1 = v1 ? xb[j1 * DM + i] : 0.0f;
        acc += x0 * w0 + x1 * w1;
    }
    size_t r = (size_t)rowi * 256;
    hS1[r + c]       = (half_t)acc;
    hS1[r + 128 + c] = (half_t)skip[(size_t)rowi * DM + c];
}

// ---------------------------------------------------------------------------
// BatchNorm stats: one block per channel, reduce over all 32768 positions
// stats[c]=mean, stats[128+c]=rsqrt(var+eps)
// ---------------------------------------------------------------------------
__global__ __launch_bounds__(256) void k_bn_stats(const float* __restrict__ F,
                                                  float* __restrict__ stats)
{
    const int c = blockIdx.x;
    float s = 0.f, s2 = 0.f;
    for (int r = threadIdx.x; r < M_ROWS; r += 256) {
        float v = F[(size_t)r * DM + c];
        s += v; s2 += v * v;
    }
    __shared__ float sh0[256], sh1[256];
    sh0[threadIdx.x] = s; sh1[threadIdx.x] = s2;
    __syncthreads();
    for (int o = 128; o > 0; o >>= 1) {
        if (threadIdx.x < o) { sh0[threadIdx.x] += sh0[threadIdx.x + o];
                               sh1[threadIdx.x] += sh1[threadIdx.x + o]; }
        __syncthreads();
    }
    if (threadIdx.x == 0) {
        float mean = sh0[0] / (float)M_ROWS;
        float var  = sh1[0] / (float)M_ROWS - mean * mean;
        stats[c]       = mean;
        stats[128 + c] = rsqrtf(var + EPSV);
    }
}

// Apply BN and stage normalized result as f16 into hS1 cols [0,128)
__global__ void k_bn_apply(const float* __restrict__ F,
                           const float* __restrict__ stats,
                           const float* __restrict__ g,
                           const float* __restrict__ b,
                           half_t* __restrict__ hS1)
{
    int idx = blockIdx.x * blockDim.x + threadIdx.x;
    if (idx >= M_ROWS * DM) return;
    int c = idx & 127;
    int rowi = idx >> 7;
    float v = (F[idx] - stats[c]) * stats[128 + c] * g[c] + b[c];
    hS1[(size_t)rowi * 256 + c] = (half_t)v;
}

// ---------------------------------------------------------------------------
// Adjacency mix over nodes + layernorm over channels; write f16 into
// hS1 cols [128,256).  block = one (b,m,t), threads = 128 channels
// ---------------------------------------------------------------------------
__global__ __launch_bounds__(128) void k_gcn(const float* __restrict__ adj,
                                             const float* __restrict__ XP,
                                             const float* __restrict__ lng,
                                             const float* __restrict__ lnb,
                                             half_t* __restrict__ hS1)
{
    const int bid = blockIdx.x;          // b*8192 + m*64 + t == output row
    const int t = bid & 63;
    const int m = (bid >> 6) & 127;
    const int b = bid >> 13;
    const int c = threadIdx.x;

    float acc = 0.f;
    const float* arow = adj + (size_t)m * 128;
    for (int n = 0; n < 128; ++n)
        acc += arow[n] * XP[(((size_t)(b * 128 + n)) * 64 + t) * DM + c];

    __shared__ float sh[128];
    sh[c] = acc; __syncthreads();
    for (int o = 64; o > 0; o >>= 1) { if (c < o) sh[c] += sh[c + o]; __syncthreads(); }
    float mean = sh[0] / 128.f;
    __syncthreads();
    float dv = acc - mean;
    sh[c] = dv * dv; __syncthreads();
    for (int o = 64; o > 0; o >>= 1) { if (c < o) sh[c] += sh[c + o]; __syncthreads(); }
    float var = sh[0] / 128.f;
    float y = dv * rsqrtf(var + EPSV) * lng[c] + lnb[c];
    hS1[(size_t)bid * 256 + 128 + c] = (half_t)y;
}

// Row layernorm (128 cols) -> f16 staging
__global__ __launch_bounds__(128) void k_rownorm(const float* __restrict__ F,
                                                 const float* __restrict__ g,
                                                 const float* __restrict__ b,
                                                 half_t* __restrict__ out)
{
    const int rowi = blockIdx.x;
    const int c = threadIdx.x;
    float v = F[(size_t)rowi * DM + c];
    __shared__ float sh[128];
    sh[c] = v; __syncthreads();
    for (int o = 64; o > 0; o >>= 1) { if (c < o) sh[c] += sh[c + o]; __syncthreads(); }
    float mean = sh[0] / 128.f;
    __syncthreads();
    float dv = v - mean;
    sh[c] = dv * dv; __syncthreads();
    for (int o = 64; o > 0; o >>= 1) { if (c < o) sh[c] += sh[c + o]; __syncthreads(); }
    float var = sh[0] / 128.f;
    out[(size_t)rowi * DM + c] = (half_t)(dv * rsqrtf(var + EPSV) * g[c] + b[c]);
}

// ---------------------------------------------------------------------------
// Depthwise causal conv (K=4, pad-left 3) + SiLU; write fp32 XM and f16 hxm
// ---------------------------------------------------------------------------
__global__ void k_conv_silu(const float* __restrict__ XZ,
                            const float* __restrict__ cw,
                            const float* __restrict__ cb,
                            float* __restrict__ XM,
                            half_t* __restrict__ hxm)
{
    int idx = blockIdx.x * blockDim.x + threadIdx.x;   // over M*256
    if (idx >= M_ROWS * DI) return;
    int c = idx & 255;
    int rowi = idx >> 8;
    int t  = rowi & 63;
    int bn = rowi >> 6;
    float acc = cb[c];
    #pragma unroll
    for (int k = 0; k < 4; ++k) {
        int tt = t + k - 3;
        if (tt >= 0)
            acc += cw[c * 4 + k] * XZ[((size_t)(bn * 64 + tt)) * 512 + c];
    }
    float s = acc / (1.0f + __expf(-acc));   // silu
    XM[idx]  = s;
    hxm[idx] = (half_t)s;
}

// ---------------------------------------------------------------------------
// Selective scan + skip + gating. block = one sequence (bn), thread = di.
// dt computed inline: softplus(dbl[0:8] . dt_w[di] + dt_b[di]).
// h[16] and A[16] live in VGPRs; B/C broadcast via LDS each step.
// Writes gated output directly as f16 for the out_proj GEMM.
// ---------------------------------------------------------------------------
__global__ __launch_bounds__(256) void k_scan(
    const float* __restrict__ DBL,    // [M,48]: 0..7 dt_in, 8..23 B, 24..39 C
    const float* __restrict__ XM,     // [M,256]
    const float* __restrict__ XZ,     // [M,512] (z at +256)
    const float* __restrict__ dt_w,   // [256,8]
    const float* __restrict__ dt_b,   // [256]
    const float* __restrict__ A_log,  // [256,16]
    const float* __restrict__ Dp,     // [256]
    half_t* __restrict__ hy)          // [M,256] f16
{
    const int bn = blockIdx.x;
    const int di = threadIdx.x;

    float A[16], h[16], wdt[8];
    #pragma unroll
    for (int s = 0; s < 16; ++s) { A[s] = -__expf(A_log[di * 16 + s]); h[s] = 0.f; }
    #pragma unroll
    for (int r = 0; r < 8; ++r) wdt[r] = dt_w[di * 8 + r];
    const float dtb = dt_b[di];
    const float Dv  = Dp[di];

    __shared__ float Bs[16], Cs[16];

    for (int t = 0; t < TT; ++t) {
        const size_t row = (size_t)bn * TT + t;
        const float* dr = DBL + row * 48;
        if (di < 16)       Bs[di]      = dr[8 + di];
        else if (di < 32)  Cs[di - 16] = dr[24 + (di - 16)];
        __syncthreads();

        float dtv = dtb;
        #pragma unroll
        for (int r = 0; r < 8; ++r) dtv += dr[r] * wdt[r];
        dtv = (dtv > 20.0f) ? dtv : log1pf(__expf(dtv));   // softplus

        const float xmv = XM[row * DI + di];
        const float dtx = dtv * xmv;
        float y = 0.f;
        #pragma unroll
        for (int s = 0; s < 16; ++s) {
            float dA = __expf(dtv * A[s]);
            h[s] = dA * h[s] + dtx * Bs[s];
            y += h[s] * Cs[s];
        }
        y += xmv * Dv;
        const float zv = XZ[row * 512 + 256 + di];
        y *= zv / (1.0f + __expf(-zv));                    // * silu(z)
        hy[row * DI + di] = (half_t)y;
        __syncthreads();
    }
}

// ---------------------------------------------------------------------------
// Host orchestration
// ---------------------------------------------------------------------------
static inline void gemm(const half_t* A, int lda, const half_t* WT,
                        const float* bias, float* C, int ldc,
                        int Npad, int K, int relu, hipStream_t s)
{
    dim3 grid(M_ROWS / 32, Npad / 16);
    k_gemm_wmma<<<grid, 32, 0, s>>>(A, lda, WT, bias, C, ldc, Npad, K, relu);
}

static inline void stage_wT(const float* src, half_t* dst, int N, int K, int Npad,
                            hipStream_t s)
{
    int total = K * Npad;
    int blocks = (total + 255) / 256;
    if (blocks > 2048) blocks = 2048;
    k_stage_wT<<<blocks, 256, 0, s>>>(src, dst, N, K, Npad);
}

static inline void tof16(const float* src, half_t* dst, size_t n, hipStream_t s)
{
    int blocks = (int)((n + 255) / 256);
    if (blocks > 4096) blocks = 4096;
    k_tof16<<<blocks, 256, 0, s>>>(src, dst, n);
}

extern "C" void kernel_launch(void* const* d_in, const int* in_sizes, int n_in,
                              void* d_out, int out_size, void* d_ws, size_t ws_size,
                              hipStream_t stream)
{
    (void)in_sizes; (void)n_in; (void)out_size; (void)ws_size;

    const float* x       = (const float*)d_in[0];
    const float* skip    = (const float*)d_in[1];
    const float* adj     = (const float*)d_in[2];
    const float* up_w    = (const float*)d_in[3];
    const float* up_b    = (const float*)d_in[4];
    const float* fuse_w  = (const float*)d_in[5];
    const float* fuse_b  = (const float*)d_in[6];
    const float* bn_g    = (const float*)d_in[7];
    const float* bn_b    = (const float*)d_in[8];
    const float* gcn_w   = (const float*)d_in[9];
    const float* gcn_b   = (const float*)d_in[10];
    const float* gln_g   = (const float*)d_in[11];
    const float* gln_b   = (const float*)d_in[12];
    const float* gf_w    = (const float*)d_in[13];
    const float* gf_b    = (const float*)d_in[14];
    const float* norm_g  = (const float*)d_in[15];
    const float* norm_b  = (const float*)d_in[16];
    const float* in_w    = (const float*)d_in[17];  // (2,512,128)
    const float* conv_w  = (const float*)d_in[18];  // (2,256,1,4)
    const float* conv_b  = (const float*)d_in[19];  // (2,256)
    const float* xp_w    = (const float*)d_in[20];  // (2,40,256)
    const float* dt_w    = (const float*)d_in[21];  // (2,256,8)
    const float* dt_b    = (const float*)d_in[22];  // (2,256)
    const float* A_log   = (const float*)d_in[23];  // (2,256,16)
    const float* Dpar    = (const float*)d_in[24];  // (2,256)
    const float* out_w   = (const float*)d_in[25];  // (2,128,256)

    float* out = (float*)d_out;                     // [32768,128] fp32

    // workspace layout
    const size_t M = M_ROWS;
    float* F0    = (float*)d_ws;              // [M,128]
    float* F1    = F0 + M * 128;              // [M,128]
    float* XZ    = F1 + M * 128;              // [M,512]
    float* XM    = XZ + M * 512;              // [M,256]
    float* DBL   = XM + M * 256;              // [M,48]
    float* STATS = DBL + M * 48;              // 256
    half_t* HS1  = (half_t*)(STATS + 256);    // [M,256] staging
    half_t* HS2  = HS1 + M * 256;             // [M,128] staging
    half_t* Wfuse = HS2 + M * 128;            // K=256, Npad=128
    half_t* Wgcn  = Wfuse + 256 * 128;        // K=128, Npad=128
    half_t* Wgf   = Wgcn + 128 * 128;         // K=256, Npad=128
    half_t* Win   = Wgf + 256 * 128;          // 2 * (K=128, Npad=512)
    half_t* Wxp   = Win + 2 * 128 * 512;      // 2 * (K=256, Npad=48)
    half_t* Wout  = Wxp + 2 * 256 * 48;       // 2 * (K=256, Npad=128)

    // weight staging: transpose to [K, Npad] f16 with zero padding
    stage_wT(fuse_w, Wfuse, 128, 256, 128, stream);
    stage_wT(gcn_w,  Wgcn,  128, 128, 128, stream);
    stage_wT(gf_w,   Wgf,   128, 256, 128, stream);
    for (int l = 0; l < 2; ++l) {
        stage_wT(in_w  + (size_t)l * 512 * 128, Win  + (size_t)l * 128 * 512, 512, 128, 512, stream);
        stage_wT(xp_w  + (size_t)l * 40 * 256,  Wxp  + (size_t)l * 256 * 48,  40,  256, 48,  stream);
        stage_wT(out_w + (size_t)l * 128 * 256, Wout + (size_t)l * 256 * 128, 128, 256, 128, stream);
    }

    // 1) upsample + stage [xu | skip] into HS1
    k_upsample_stage<<<(M_ROWS * DM + 255) / 256, 256, 0, stream>>>(x, skip, up_w, up_b, HS1);

    // 2) fused = [xu|skip] @ fuse_w^T + fuse_b
    gemm(HS1, 256, Wfuse, fuse_b, F0, 128, 128, 256, 0, stream);

    // 3) batchnorm over (bn,t) per channel
    k_bn_stats<<<128, 256, 0, stream>>>(F0, STATS);
    k_bn_apply<<<(M_ROWS * DM + 255) / 256, 256, 0, stream>>>(F0, STATS, bn_g, bn_b, HS1);

    // 4) GCN: xp = fused @ gcn_w^T + b   (A = HS1 cols[0,128), lda=256)
    gemm(HS1, 256, Wgcn, gcn_b, F1, 128, 128, 128, 0, stream);
    //    adjacency mix + layernorm -> HS1 cols[128,256)
    k_gcn<<<M_ROWS, 128, 0, stream>>>(adj, F1, gln_g, gln_b, HS1);

    // 5) gated fusion: relu([fused|go] @ gf_w^T + b), then row layernorm -> HS2
    gemm(HS1, 256, Wgf, gf_b, F0, 128, 128, 256, 1, stream);
    k_rownorm<<<M_ROWS, 128, 0, stream>>>(F0, norm_g, norm_b, HS2);

    // 6) Mamba layers
    for (int l = 0; l < 2; ++l) {
        // xz = u @ in_proj^T
        gemm(HS2, 128, Win + (size_t)l * 128 * 512, nullptr, XZ, 512, 512, 128, 0, stream);
        // depthwise causal conv + silu
        k_conv_silu<<<(M_ROWS * DI + 255) / 256, 256, 0, stream>>>(
            XZ, conv_w + (size_t)l * 256 * 4, conv_b + (size_t)l * 256, XM, HS1);
        // dbl = xm @ x_proj^T  (N=40 padded to 48, ldc=48; pad cols compute 0)
        gemm(HS1, 256, Wxp + (size_t)l * 256 * 48, nullptr, DBL, 48, 48, 256, 0, stream);
        // selective scan + gating -> HS1 (f16)
        k_scan<<<BN, DI, 0, stream>>>(DBL, XM, XZ,
                                      dt_w + (size_t)l * 256 * 8,
                                      dt_b + (size_t)l * 256,
                                      A_log + (size_t)l * 256 * 16,
                                      Dpar + (size_t)l * 256, HS1);
        // out = y @ out_proj^T
        float* dst = (l == 0) ? F0 : out;
        gemm(HS1, 256, Wout + (size_t)l * 256 * 128, nullptr, dst, 128, 128, 256, 0, stream);
        if (l == 0) tof16(F0, HS2, M * 128, stream);
    }
}
